// TopKRouter_83837761618192
// MI455X (gfx1250) — matrix operand order repaired and checked by reading, not compiled, not run on hardware
//
#include <hip/hip_runtime.h>
#include <hip/hip_bf16.h>

// ---- vector types matching gfx1250 WMMA builtin signatures ----
typedef __bf16        v16bf __attribute__((ext_vector_type(16)));
typedef float         v8f   __attribute__((ext_vector_type(8)));
typedef float         f32x4 __attribute__((ext_vector_type(4)));
typedef unsigned int  u32x4 __attribute__((ext_vector_type(4)));

#define EXPERTS         64
#define TILE_M          16
#define WAVES_PER_BLOCK 8
#define LDS_STRIDE      66   // 64 + 2 pad: avoids LDS bank conflicts on row stores

// ---- pre-kernel: W_gate fp32 (E x D) -> bf16, held in d_ws (stays hot in L2) ----
__global__ __launch_bounds__(256) void convert_w_kernel(const float* __restrict__ W,
                                                        __bf16* __restrict__ Wb, int n) {
  int i = blockIdx.x * blockDim.x + threadIdx.x;
  if (i < n) Wb[i] = (__bf16)W[i];
}

// ---- main: one wave = 16 tokens x 64 experts, K-loop of 32 via v_wmma_f32_16x16x32_bf16 ----
__global__ __launch_bounds__(256) void router_kernel(const float* __restrict__ x,
                                                     const __bf16* __restrict__ Wb,
                                                     float* __restrict__ out,
                                                     int D, int tokens) {
  __shared__ float lds[WAVES_PER_BLOCK * TILE_M * LDS_STRIDE];

  const int lane = threadIdx.x & 31;
  const int wave = threadIdx.x >> 5;
  const int tile = blockIdx.x * WAVES_PER_BLOCK + wave;
  const int row_base = tile * TILE_M;

  const int en = lane & 15;   // row (A) / expert column (B) owned by this lane
  const int hs = lane >> 4;   // which K-half this lane holds

  // A: lane L and L+16 hold the same token row, different K-subblocks
  const float* xrow = x + (size_t)(row_base + en) * D;
  const int kA = hs * 16;     // lane<16: K 0..15; lane>=16: K 16..31

  // B: lane = expert column; lane<16 holds K 0..15, lane>=16 holds K 16..31 (contiguous!)
  const __bf16* wrow[4];
#pragma unroll
  for (int t = 0; t < 4; ++t)
    wrow[t] = Wb + (size_t)(t * 16 + en) * D + hs * 16;

  v8f acc[4] = {};  // 4 accumulators: experts [0..15],[16..31],[32..47],[48..63]

  for (int k0 = 0; k0 < D; k0 += 32) {
    // ---- A tile: 16 fp32 per lane (one contiguous 16-float block), NT so streaming
    //      x doesn't evict W's bf16 copy from L2; convert to bf16 in registers.
    const float* pa = xrow + k0 + kA;
    f32x4 xf[4];
    xf[0] = __builtin_nontemporal_load((const f32x4*)(pa));
    xf[1] = __builtin_nontemporal_load((const f32x4*)(pa + 4));
    xf[2] = __builtin_nontemporal_load((const f32x4*)(pa + 8));
    xf[3] = __builtin_nontemporal_load((const f32x4*)(pa + 12));
    v16bf a;
#pragma unroll
    for (int j = 0; j < 16; ++j) a[j] = (__bf16)xf[j >> 2][j & 3];

    // ---- 4 expert tiles: B already bf16, 16 contiguous values per lane (2x b128)
#pragma unroll
    for (int t = 0; t < 4; ++t) {
      union { v16bf v; u32x4 u[2]; } b;
      const u32x4* pb = (const u32x4*)(wrow[t] + k0);
      b.u[0] = pb[0];
      b.u[1] = pb[1];
      acc[t] = __builtin_amdgcn_wmma_f32_16x16x32_bf16(
          /*neg_a=*/false, a, /*neg_b=*/false, b.v,
          /*c_mod=*/(short)0, acc[t], /*reuse_a=*/false, /*reuse_b=*/false);
    }
  }

  // ---- spill 16x64 logits to LDS (C/D layout: VGPR r -> M = r + 8*hs, lane -> N)
  float* lbase = lds + wave * TILE_M * LDS_STRIDE;
#pragma unroll
  for (int t = 0; t < 4; ++t)
#pragma unroll
    for (int r = 0; r < 8; ++r)
      lbase[(r + 8 * hs) * LDS_STRIDE + 16 * t + en] = acc[t][r];

  __syncthreads();

  const size_t w_off     = (size_t)tokens * 2;
  const size_t logit_off = (size_t)tokens * 4;

  // ---- per-token softmax + top-2 (top-2 of logits == top-2 of probs)
  if (lane < 16) {
    const float* rowl = lbase + lane * LDS_STRIDE;
    float mx = rowl[0];
    for (int e = 1; e < EXPERTS; ++e) mx = fmaxf(mx, rowl[e]);
    float s = 0.0f, l0 = -3.4e38f, l1 = -3.4e38f;
    int i0 = 0, i1 = 0;
    for (int e = 0; e < EXPERTS; ++e) {
      float v = rowl[e];
      s += __expf(v - mx);
      if (v > l0)      { l1 = l0; i1 = i0; l0 = v; i0 = e; }
      else if (v > l1) { l1 = v; i1 = e; }
    }
    // weights = p_j / (p0 + p1 + 1e-9), p_j = exp(l_j - mx)/s
    float e0 = __expf(l0 - mx), e1 = __expf(l1 - mx);
    float denom = e0 + e1 + 1e-9f * s;
    int tok = row_base + lane;
    out[(size_t)tok * 2 + 0]         = (float)i0;
    out[(size_t)tok * 2 + 1]         = (float)i1;
    out[w_off + (size_t)tok * 2 + 0] = e0 / denom;
    out[w_off + (size_t)tok * 2 + 1] = e1 / denom;
  }

  // ---- coalesced logits write-out from LDS
  for (int i = lane; i < TILE_M * EXPERTS; i += 32) {
    int m = i >> 6, e = i & 63;
    out[logit_off + (size_t)(row_base + m) * EXPERTS + e] =
        lbase[m * LDS_STRIDE + e];
  }
}

extern "C" void kernel_launch(void* const* d_in, const int* in_sizes, int n_in,
                              void* d_out, int out_size, void* d_ws, size_t ws_size,
                              hipStream_t stream) {
  const float* x = (const float*)d_in[0];
  const float* W = (const float*)d_in[1];
  float* out = (float*)d_out;

  const int nW = in_sizes[1];            // E*D = 64*4096
  const int D = nW / EXPERTS;            // 4096
  const int tokens = in_sizes[0] / D;    // B*S = 16384

  __bf16* Wb = (__bf16*)d_ws;            // 512 KB bf16 copy of W_gate

  convert_w_kernel<<<(nW + 255) / 256, 256, 0, stream>>>(W, Wb, nW);

  const int tiles = tokens / TILE_M;             // 1024 waves
  const int blocks = tiles / WAVES_PER_BLOCK;    // 128 blocks of 8 waves
  router_kernel<<<blocks, 256, 0, stream>>>(x, Wb, out, D, tokens);
}
